// InteractionBlockWithCls_32272384262575
// MI455X (gfx1250) — matrix operand order repaired and checked by reading, not compiled, not run on hardware
//
#include <hip/hip_runtime.h>
#include <hip/hip_bf16.h>
#include <math.h>

// ---------------------------------------------------------------------------
// Problem constants (from the reference: D=768, NH=6, NP=4, B=4, H=W=32)
// ---------------------------------------------------------------------------
#define DM   768
#define NHD  6
#define HDIM 128
#define NPTS 4
#define BATCH 4
#define LQ   1024          // HH*WW
#define LC   5376          // 21*(HH*WW)/4
#define MX   (BATCH*LQ)    // 4096  rows of x
#define MC   (BATCH*LC)    // 21504 rows of c / c24
#define CHID 192           // D/4 hidden of conv-FFN

typedef __bf16 v16bf __attribute__((ext_vector_type(16)));
typedef __bf16 v4bf  __attribute__((ext_vector_type(4)));
typedef float  v8f   __attribute__((ext_vector_type(8)));

__device__ __forceinline__ __bf16 f2bf(float f) {
  union { float f; unsigned u; } a; a.f = f;
  unsigned r = (a.u + 0x7FFFu + ((a.u >> 16) & 1u)) >> 16;   // round-to-nearest-even
  union { unsigned short s; __bf16 b; } o; o.s = (unsigned short)r;
  return o.b;
}

// ---------------------------------------------------------------------------
// WMMA GEMM:  C[M,N] = A[M,K] @ W[K,N] + bias[N]   (fp32 in/out, bf16 compute)
//
// Block: 256 threads = 8 waves as a 2x4 wave grid. Block tile 64x128; each
// wave owns a 32x32 output patch = 2x2 WMMA tiles -> 4 v_wmma per K-step.
// Software-pipelined with double-buffered LDS: global loads for tile k+1 are
// issued before the WMMAs on tile k, conversion+LDS store happens after, and
// a single barrier per K-step separates the stages. LDS tiles are staged
// pre-swizzled into the exact v_wmma_f32_16x16x32_bf16 fragment layout
// (ISA 7.12.2), so fragment loads are contiguous ds_load_b128 pairs.
// Requires: K % 32 == 0, M % 64 == 0, N % 4 == 0 (true for every GEMM here).
// ---------------------------------------------------------------------------
__global__ __launch_bounds__(256) void k_gemm_bias(
    const float* __restrict__ A, const float* __restrict__ W,
    const float* __restrict__ bias, float* __restrict__ C,
    int M, int N, int K)
{
  __shared__ __align__(16) __bf16 As[2][4][512];   // [buf][16-row block][frag]
  __shared__ __align__(16) __bf16 Bs[2][8][512];   // [buf][16-col block][frag]

  const int tid  = threadIdx.x;
  const int lane = tid & 31;
  const int wave = tid >> 5;
  const int wr   = wave >> 2;        // 0..1 : 32-row strip
  const int wc   = wave & 3;         // 0..3 : 32-col strip
  const int row0 = blockIdx.y * 64;
  const int col0 = blockIdx.x * 128;
  const bool fullN = (col0 + 128 <= N);   // block-uniform: no column guards needed

  // ---- K-invariant staging coordinates -------------------------------------
  // A: 2 float4 quads per thread out of 512 (64 rows x 32 K)
  int ar[2], akq[2], aoff[2];
#pragma unroll
  for (int t = 0; t < 2; ++t) {
    int qi = tid + t * 256;
    ar[t]  = qi >> 3;                       // 0..63
    akq[t] = (qi & 7) << 2;                 // 0,4,...,28
    int l  = (ar[t] & 15) + (((akq[t] & 15) >= 8) ? 16 : 0);
    int e0 = (akq[t] & 7) + ((akq[t] >= 16) ? 8 : 0);
    aoff[t] = (ar[t] >> 4) * 512 + l * 16 + e0;     // flat offset in As[buf]
  }
  // B: 4 float4 quads per thread out of 1024 (32 K x 128 cols)
  int bkk[4], bnq[4], boff[4];
#pragma unroll
  for (int t = 0; t < 4; ++t) {
    int qi = tid + t * 256;
    bkk[t] = qi >> 5;                       // 0..31
    bnq[t] = (qi & 31) << 2;                // 0,4,...,124
    int cb = bnq[t] >> 4;
    int lb = (bnq[t] & 15) + (((bkk[t] & 15) >= 8) ? 16 : 0);
    boff[t] = cb * 512 + lb * 16 + (bkk[t] & 7) + ((bkk[t] >= 16) ? 8 : 0);   // flat offset in Bs[buf]
  }

  v8f acc00 = {}, acc01 = {}, acc10 = {}, acc11 = {};
  float4 a4[2], b4[4];

  auto loadTiles = [&](int k0) {
#pragma unroll
    for (int t = 0; t < 2; ++t)
      a4[t] = *(const float4*)&A[(size_t)(row0 + ar[t]) * K + (k0 + akq[t])];
    if (fullN) {
#pragma unroll
      for (int t = 0; t < 4; ++t)
        b4[t] = *(const float4*)&W[(size_t)(k0 + bkk[t]) * N + (col0 + bnq[t])];
    } else {
#pragma unroll
      for (int t = 0; t < 4; ++t) {
        int gn = col0 + bnq[t];
        const float* wp = &W[(size_t)(k0 + bkk[t]) * N];
        float4 v;
        if (gn + 3 < N) {
          v = *(const float4*)&wp[gn];
        } else {
          v.x = (gn + 0 < N) ? wp[gn + 0] : 0.f;
          v.y = (gn + 1 < N) ? wp[gn + 1] : 0.f;
          v.z = (gn + 2 < N) ? wp[gn + 2] : 0.f;
          v.w = (gn + 3 < N) ? wp[gn + 3] : 0.f;
        }
        b4[t] = v;
      }
    }
  };
  auto storeTiles = [&](int p) {
    __bf16* ap = &As[p][0][0];
    __bf16* bp = &Bs[p][0][0];
#pragma unroll
    for (int t = 0; t < 2; ++t) {
      v4bf w4; w4.x = f2bf(a4[t].x); w4.y = f2bf(a4[t].y);
      w4.z = f2bf(a4[t].z); w4.w = f2bf(a4[t].w);
      *(v4bf*)&ap[aoff[t]] = w4;            // one 8-byte LDS store
    }
#pragma unroll
    for (int t = 0; t < 4; ++t) {
      bp[boff[t] +  0] = f2bf(b4[t].x);
      bp[boff[t] + 16] = f2bf(b4[t].y);
      bp[boff[t] + 32] = f2bf(b4[t].z);
      bp[boff[t] + 48] = f2bf(b4[t].w);
    }
  };
  auto compute = [&](int p) {
    const v16bf fa0 = *(const v16bf*)&As[p][2 * wr + 0][lane * 16];
    const v16bf fa1 = *(const v16bf*)&As[p][2 * wr + 1][lane * 16];
    const v16bf fb0 = *(const v16bf*)&Bs[p][2 * wc + 0][lane * 16];
    const v16bf fb1 = *(const v16bf*)&Bs[p][2 * wc + 1][lane * 16];
    acc00 = __builtin_amdgcn_wmma_f32_16x16x32_bf16(false, fa0, false, fb0,
                                                    (short)0, acc00, false, false);
    acc01 = __builtin_amdgcn_wmma_f32_16x16x32_bf16(false, fa0, false, fb1,
                                                    (short)0, acc01, false, false);
    acc10 = __builtin_amdgcn_wmma_f32_16x16x32_bf16(false, fa1, false, fb0,
                                                    (short)0, acc10, false, false);
    acc11 = __builtin_amdgcn_wmma_f32_16x16x32_bf16(false, fa1, false, fb1,
                                                    (short)0, acc11, false, false);
  };

  // ---- software pipeline ---------------------------------------------------
  loadTiles(0);
  storeTiles(0);
  __syncthreads();
  int p = 0;
  for (int k0 = 32; k0 < K; k0 += 32) {
    loadTiles(k0);       // global loads in flight ...
    compute(p);          // ... while WMMAs run on the staged tile
    storeTiles(p ^ 1);   // convert + stage into the other buffer
    __syncthreads();
    p ^= 1;
  }
  compute(p);

  // ---- store: VGPR i -> row (i + 8*(lane>=16)), col = lane&15 --------------
  const int lcol = lane & 15;
  const int lrow = (lane >> 4) << 3;
#pragma unroll
  for (int i = 0; i < 2; ++i) {
#pragma unroll
    for (int j = 0; j < 2; ++j) {
      int n = col0 + wc * 32 + j * 16 + lcol;
      if (fullN || n < N) {
        float bv = bias ? bias[n] : 0.0f;
        int mbase = row0 + wr * 32 + i * 16 + lrow;
        const v8f* accp = (i == 0) ? ((j == 0) ? &acc00 : &acc01)
                                   : ((j == 0) ? &acc10 : &acc11);
#pragma unroll
        for (int r = 0; r < 8; ++r)
          C[(size_t)(mbase + r) * N + n] = (*accp)[r] + bv;
      }
    }
  }
}

// ---------------------------------------------------------------------------
// LayerNorm: one block (256 thr = 8 waves) per row of length D=768
// ---------------------------------------------------------------------------
__global__ __launch_bounds__(256) void k_ln(
    const float* __restrict__ x, const float* __restrict__ g,
    const float* __restrict__ b, float* __restrict__ y, int Dn)
{
  const int row = blockIdx.x;
  const float* xr = x + (size_t)row * Dn;
  float* yr = y + (size_t)row * Dn;
  __shared__ float sred[8];
  __shared__ float smean, srstd;
  const int tid = threadIdx.x, lane = tid & 31, wv = tid >> 5;

  float s = 0.f;
  for (int i = tid; i < Dn; i += 256) s += xr[i];
  for (int o = 16; o > 0; o >>= 1) s += __shfl_down(s, o);
  if (lane == 0) sred[wv] = s;
  __syncthreads();
  if (tid == 0) { float t = 0.f; for (int i = 0; i < 8; ++i) t += sred[i]; smean = t / Dn; }
  __syncthreads();
  const float mean = smean;

  float v = 0.f;
  for (int i = tid; i < Dn; i += 256) { float d = xr[i] - mean; v += d * d; }
  for (int o = 16; o > 0; o >>= 1) v += __shfl_down(v, o);
  __syncthreads();
  if (lane == 0) sred[wv] = v;
  __syncthreads();
  if (tid == 0) { float t = 0.f; for (int i = 0; i < 8; ++i) t += sred[i];
                  srstd = rsqrtf(t / Dn + 1e-6f); }
  __syncthreads();
  const float rs = srstd;
  for (int i = tid; i < Dn; i += 256)
    yr[i] = (xr[i] - mean) * rs * g[i] + b[i];
}

// ---------------------------------------------------------------------------
// MSDA sampling: block per (b, head, query), 128 threads = one hd channel each.
// Fuses softmax over the L*NP (<=12) attention logits.
// V:(B,Lv,768)  off:(B,Lq,NH,L,NP,2)  awl:(B,Lq,NH,L*NP)  ref:(B,Lq,1,2)
// out:(B,Lq,768) laid out as (B,Lq,NH,hd)
// ---------------------------------------------------------------------------
__global__ __launch_bounds__(128) void k_msda(
    const float* __restrict__ V, const float* __restrict__ off,
    const float* __restrict__ awl, const float* __restrict__ ref,
    float* __restrict__ out, int Lq, int Lv, int L,
    int H0, int W0, int S0, int H1, int W1, int S1, int H2, int W2, int S2)
{
  const int gid = blockIdx.x;
  const int q = gid % Lq;
  const int h = (gid / Lq) % NHD;
  const int b = gid / (Lq * NHD);
  const int d = threadIdx.x;
  const int NPL = L * NPTS;

  const float* aw = awl + (((size_t)b * Lq + q) * NHD + h) * NPL;
  float mx = -1e30f;
  for (int i = 0; i < NPL; ++i) mx = fmaxf(mx, aw[i]);
  float wts[12]; float se = 0.f;
  for (int i = 0; i < NPL; ++i) { wts[i] = __expf(aw[i] - mx); se += wts[i]; }
  const float inv = 1.f / se;

  const float* of = off + (((size_t)b * Lq + q) * NHD + h) * (size_t)L * NPTS * 2;
  const float rx = ref[((size_t)b * Lq + q) * 2 + 0];
  const float ry = ref[((size_t)b * Lq + q) * 2 + 1];
  const float* Vb = V + (size_t)b * Lv * DM + h * HDIM + d;

  int Hs[3] = {H0, H1, H2}, Ws[3] = {W0, W1, W2}, Ss[3] = {S0, S1, S2};

  float accv = 0.f;
  for (int l = 0; l < L; ++l) {
    const int Hl = Hs[l], Wl = Ws[l], base = Ss[l];
    for (int p = 0; p < NPTS; ++p) {
      const float ofx = of[(l * NPTS + p) * 2 + 0];
      const float ofy = of[(l * NPTS + p) * 2 + 1];
      const float xp = rx * Wl + ofx - 0.5f;   // (ref + off/norm)*W - 0.5
      const float yp = ry * Hl + ofy - 0.5f;
      const float x0f = floorf(xp), y0f = floorf(yp);
      const int x0 = (int)x0f, y0 = (int)y0f;
      const float wx = xp - x0f, wy = yp - y0f;
      const float a = wts[l * NPTS + p] * inv;
      float sv = 0.f;
#pragma unroll
      for (int dy = 0; dy < 2; ++dy)
#pragma unroll
        for (int dx = 0; dx < 2; ++dx) {
          const int xi = x0 + dx, yi = y0 + dy;
          const float wgt = (dx ? wx : 1.f - wx) * (dy ? wy : 1.f - wy);
          if (xi >= 0 && xi < Wl && yi >= 0 && yi < Hl)
            sv += wgt * Vb[(size_t)(base + yi * Wl + xi) * DM];
        }
      accv += a * sv;
    }
  }
  out[((size_t)b * Lq + q) * DM + h * HDIM + d] = accv;
}

// ---------------------------------------------------------------------------
// Depthwise 3x3 SAME conv on one pyramid level segment (NHWC, groups=C)
// in/out: (B, rowsPerBatch, C) with the level at rows [seg0, seg0+Hl*Wl)
// w: (3,3,1,C)
// ---------------------------------------------------------------------------
__global__ __launch_bounds__(256) void k_dwconv(
    const float* __restrict__ in, const float* __restrict__ w,
    const float* __restrict__ bias, float* __restrict__ outp,
    int rowsPerBatch, int seg0, int Hl, int Wl, int C)
{
  size_t idx = (size_t)blockIdx.x * 256 + threadIdx.x;
  size_t tot = (size_t)BATCH * Hl * Wl * C;
  if (idx >= tot) return;
  int ch  = (int)(idx % C);
  int pos = (int)((idx / C) % ((size_t)Hl * Wl));
  int b   = (int)(idx / ((size_t)C * Hl * Wl));
  int y = pos / Wl, x = pos % Wl;
  const float* inb = in + ((size_t)b * rowsPerBatch + seg0) * C;
  float s = bias[ch];
#pragma unroll
  for (int kh = 0; kh < 3; ++kh) {
    int yy = y + kh - 1; if (yy < 0 || yy >= Hl) continue;
#pragma unroll
    for (int kw = 0; kw < 3; ++kw) {
      int xx = x + kw - 1; if (xx < 0 || xx >= Wl) continue;
      s += w[(kh * 3 + kw) * C + ch] * inb[(size_t)(yy * Wl + xx) * C + ch];
    }
  }
  outp[((size_t)b * rowsPerBatch + seg0 + pos) * C + ch] = s;
}

// ---------------------------------------------------------------------------
// Elementwise helpers
// ---------------------------------------------------------------------------
__global__ __launch_bounds__(256) void k_gelu(float* __restrict__ t, size_t n) {
  size_t i = (size_t)blockIdx.x * 256 + threadIdx.x;
  if (i >= n) return;
  float v = t[i];
  t[i] = 0.5f * v * (1.0f + erff(v * 0.70710678118654752f));
}

__global__ __launch_bounds__(256) void k_axpy(
    float* __restrict__ o, const float* __restrict__ t,
    const float* __restrict__ gamma, size_t n, int Dn) {
  size_t i = (size_t)blockIdx.x * 256 + threadIdx.x;
  if (i >= n) return;
  float g = gamma ? gamma[i % Dn] : 1.0f;
  o[i] += g * t[i];
}

__global__ __launch_bounds__(256) void k_copy(
    float* __restrict__ o, const float* __restrict__ t, size_t n) {
  size_t i = (size_t)blockIdx.x * 256 + threadIdx.x;
  if (i < n) o[i] = t[i];
}

// ---------------------------------------------------------------------------
// Host side: full forward.  Input order = setup_inputs() with the params dict
// flattened in jax pytree (sorted-key) order:
//   6..43  params['ext']  (attn, attn_o, ffn, ffn_o, ffnn_*, ffnno_*, fn_*, qn_*, qno_*)
//   44..69 params['inj']  (attn, attn_o, fn_*, fno_*, gamma, gamma_o, qn_*, qno_*)
//   attn sub-dict order: ab, aw, ob, outb, outw, ow, vb, vw
//   ffn  sub-dict order: dwb, dww, fc1b, fc1w, fc2b, fc2w
// ---------------------------------------------------------------------------
extern "C" void kernel_launch(void* const* d_in, const int* in_sizes, int n_in,
                              void* d_out, int out_size, void* d_ws, size_t ws_size,
                              hipStream_t stream) {
  (void)in_sizes; (void)n_in; (void)out_size; (void)ws_size;
  const float* x_in   = (const float*)d_in[0];
  const float* c_in   = (const float*)d_in[1];
  const float* c24_in = (const float*)d_in[2];
  const float* cls_in = (const float*)d_in[3];
  const float* ref1   = (const float*)d_in[4];
  const float* ref2   = (const float*)d_in[5];
#define P(i) ((const float*)d_in[(i)])

  float* out  = (float*)d_out;
  float* ox   = out;                       // (4,1024,768)
  float* oc   = ox   + (size_t)MX * DM;    // (4,5376,768)
  float* oc24 = oc   + (size_t)MC * DM;
  float* ocls = oc24 + (size_t)MC * DM;    // (4,1,768)

  float* ws   = (float*)d_ws;
  float* Bln  = ws;                          // 21504*768
  float* Bv   = Bln  + (size_t)MC * DM;      // 21504*768
  float* Bms  = Bv   + (size_t)MC * DM;      // 21504*768
  float* Bv2  = Bms  + (size_t)MC * DM;      // 4096*768
  float* Boff = Bv2  + (size_t)MX * DM;      // up to 21504*144
  float* Baw  = Boff + (size_t)MC * 144;     // up to 21504*72
  float* Bfc1 = Baw  + (size_t)MC * 72;      // 21504*192
  float* Bdw  = Bfc1 + (size_t)MC * CHID;    // 21504*192

  auto gemm = [&](const float* A, const float* Wm, const float* bias, float* C,
                  int M, int N, int K) {
    dim3 g((N + 127) / 128, M / 64);
    k_gemm_bias<<<g, dim3(256), 0, stream>>>(A, Wm, bias, C, M, N, K);
  };
  auto ln = [&](const float* xx, const float* g, const float* bb, float* y, int rows) {
    k_ln<<<dim3(rows), dim3(256), 0, stream>>>(xx, g, bb, y, DM);
  };
  auto msda3 = [&](const float* V, const float* offb, const float* awb,
                   const float* refp, float* o, int Lq_, int Lv_) {
    k_msda<<<dim3(BATCH * NHD * Lq_), dim3(128), 0, stream>>>(
        V, offb, awb, refp, o, Lq_, Lv_, 3, 64, 64, 0, 32, 32, 4096, 16, 16, 5120);
  };
  auto msda1 = [&](const float* V, const float* offb, const float* awb,
                   const float* refp, float* o, int Lq_, int Lv_) {
    k_msda<<<dim3(BATCH * NHD * Lq_), dim3(128), 0, stream>>>(
        V, offb, awb, refp, o, Lq_, Lv_, 1, 32, 32, 0, 0, 0, 0, 0, 0, 0);
  };
  auto axpy = [&](float* o, const float* t, const float* g, size_t n) {
    k_axpy<<<dim3((unsigned)((n + 255) / 256)), dim3(256), 0, stream>>>(o, t, g, n, DM);
  };
  auto copyf = [&](float* o, const float* t, size_t n) {
    k_copy<<<dim3((unsigned)((n + 255) / 256)), dim3(256), 0, stream>>>(o, t, n);
  };

  const size_t NXE = (size_t)MX * DM, NCE = (size_t)MC * DM;

  // residual bases
  copyf(ox, x_in, NXE);
  copyf(oc, c_in, NCE);
  copyf(oc24, c24_in, NCE);
  copyf(ocls, cls_in, (size_t)BATCH * DM);

  // ================= inj: x = x + gamma * MSDA(LN(x), ref1, LN(c)) ==========
  ln(c_in, P(61), P(60), Bln, MC);                       // fn = LN(c)
  gemm(Bln, P(51), P(50), Bv, MC, DM, DM);               // value
  ln(ox, P(67), P(66), Bln, MX);                         // qn = LN(x)
  gemm(Bln, P(49), P(46), Boff, MX, 144, DM);            // offsets
  gemm(Bln, P(45), P(44), Baw, MX, 72, DM);              // attn logits
  msda3(Bv, Boff, Baw, ref1, Bms, LQ, LC);               // sampling
  gemm(Bms, P(48), P(47), Bln, MX, DM, DM);              // out proj
  axpy(ox, Bln, P(64), NXE);                             // x += gamma * ...

  // ============ inj(attn_o): x = x + gamma_o * MSDA(LN(x), ref1, LN(c24)) ===
  ln(c24_in, P(63), P(62), Bln, MC);                     // fno = LN(c24)
  gemm(Bln, P(59), P(58), Bv, MC, DM, DM);
  ln(ox, P(69), P(68), Bln, MX);                         // qno = LN(x updated)
  gemm(Bln, P(57), P(54), Boff, MX, 144, DM);
  gemm(Bln, P(53), P(52), Baw, MX, 72, DM);
  msda3(Bv, Boff, Baw, ref1, Bms, LQ, LC);
  gemm(Bms, P(56), P(55), Bln, MX, DM, DM);
  axpy(ox, Bln, P(65), NXE);

  // ================= ext: fn = LN(x); two value projections =================
  ln(ox, P(39), P(38), Bln, MX);                         // fn = LN(x)
  gemm(Bln, P(13), P(12), Bv,  MX, DM, DM);              // value (attn)
  gemm(Bln, P(21), P(20), Bv2, MX, DM, DM);              // value (attn_o)

  // ext.attn: c += MSDA(LN(c), ref2, fn)
  ln(c_in, P(41), P(40), Bln, MC);                       // qn = LN(c)
  gemm(Bln, P(11), P(8), Boff, MC, 48, DM);
  gemm(Bln, P(7),  P(6), Baw,  MC, 24, DM);
  msda1(Bv, Boff, Baw, ref2, Bms, LC, LQ);
  gemm(Bms, P(10), P(9), Bln, MC, DM, DM);
  axpy(oc, Bln, nullptr, NCE);

  // ext.attn_o: c24 += MSDA(LN(c24), ref2, fn)
  ln(c24_in, P(43), P(42), Bln, MC);                     // qno = LN(c24)
  gemm(Bln, P(19), P(16), Boff, MC, 48, DM);
  gemm(Bln, P(15), P(14), Baw,  MC, 24, DM);
  msda1(Bv2, Boff, Baw, ref2, Bms, LC, LQ);
  gemm(Bms, P(18), P(17), Bln, MC, DM, DM);
  axpy(oc24, Bln, nullptr, NCE);

  // ================= conv-FFN on c ==========================================
  const size_t NH192 = (size_t)MC * CHID;
  ln(oc, P(35), P(34), Bln, MC);
  gemm(Bln, P(25), P(24), Bfc1, MC, CHID, DM);           // fc1
  {
    size_t t0 = (size_t)BATCH * 64 * 64 * CHID;
    size_t t1 = (size_t)BATCH * 32 * 32 * CHID;
    size_t t2 = (size_t)BATCH * 16 * 16 * CHID;
    k_dwconv<<<dim3((unsigned)((t0 + 255) / 256)), dim3(256), 0, stream>>>(
        Bfc1, P(23), P(22), Bdw, LC, 0,    64, 64, CHID);
    k_dwconv<<<dim3((unsigned)((t1 + 255) / 256)), dim3(256), 0, stream>>>(
        Bfc1, P(23), P(22), Bdw, LC, 4096, 32, 32, CHID);
    k_dwconv<<<dim3((unsigned)((t2 + 255) / 256)), dim3(256), 0, stream>>>(
        Bfc1, P(23), P(22), Bdw, LC, 5120, 16, 16, CHID);
  }
  k_gelu<<<dim3((unsigned)((NH192 + 255) / 256)), dim3(256), 0, stream>>>(Bdw, NH192);
  gemm(Bdw, P(27), P(26), Bms, MC, DM, CHID);            // fc2
  axpy(oc, Bms, nullptr, NCE);

  // ================= conv-FFN on c24 ========================================
  ln(oc24, P(37), P(36), Bln, MC);
  gemm(Bln, P(31), P(30), Bfc1, MC, CHID, DM);
  {
    size_t t0 = (size_t)BATCH * 64 * 64 * CHID;
    size_t t1 = (size_t)BATCH * 32 * 32 * CHID;
    size_t t2 = (size_t)BATCH * 16 * 16 * CHID;
    k_dwconv<<<dim3((unsigned)((t0 + 255) / 256)), dim3(256), 0, stream>>>(
        Bfc1, P(29), P(28), Bdw, LC, 0,    64, 64, CHID);
    k_dwconv<<<dim3((unsigned)((t1 + 255) / 256)), dim3(256), 0, stream>>>(
        Bfc1, P(29), P(28), Bdw, LC, 4096, 32, 32, CHID);
    k_dwconv<<<dim3((unsigned)((t2 + 255) / 256)), dim3(256), 0, stream>>>(
        Bfc1, P(29), P(28), Bdw, LC, 5120, 16, 16, CHID);
  }
  k_gelu<<<dim3((unsigned)((NH192 + 255) / 256)), dim3(256), 0, stream>>>(Bdw, NH192);
  gemm(Bdw, P(33), P(32), Bms, MC, DM, CHID);
  axpy(oc24, Bms, nullptr, NCE);
#undef P
}
